// UncertainNN_61306363183460
// MI455X (gfx1250) — matrix operand order repaired and checked by reading, not compiled, not run on hardware
//
#include <hip/hip_runtime.h>
#include <hip/hip_bf16.h>

typedef __attribute__((ext_vector_type(16))) __bf16 v16bf;
typedef __attribute__((ext_vector_type(8)))  float  v8f;

#define S_LEN 1024
#define D_MODEL 768
#define D_IN 1536
#define VOCAB 50257
#define EPSF 1e-5f

union Frag16 { v16bf v; unsigned short u[16]; };

__device__ __forceinline__ unsigned short f2bf(float f) {
  unsigned int u = __float_as_uint(f);
  u += 0x7FFFu + ((u >> 16) & 1u);
  return (unsigned short)(u >> 16);
}
__device__ __forceinline__ float siluf(float x) { return x / (1.f + __expf(-x)); }
__device__ __forceinline__ float geluf(float x) {
  float x3 = x * x * x;
  return 0.5f * x * (1.f + tanhf(0.7978845608028654f * (x + 0.044715f * x3)));
}
__device__ __forceinline__ float softplusf(float x) {
  return (x > 20.f) ? x : log1pf(__expf(x));
}

// ---------------- Strided / batched bf16 WMMA GEMM ----------------
// C[b](m,n) = act(scale * sum_k A[b](m,k)*B[b](k,n) + bias[n]) (+ C_old if accum)
// Tile: 64x64 output, BK=64, 128 threads (4 waves), double-buffered LDS.
__global__ __launch_bounds__(128) void gemm_wmma_kernel(
    const float* __restrict__ A, const float* __restrict__ B, float* __restrict__ C,
    int M, int N, int K,
    long long a_rs, long long a_cs, long long a_bs,
    long long b_rs, long long b_cs, long long b_bs,
    long long c_rs, long long c_bs,
    const float* __restrict__ bias, float scale, int act, int accum)
{
  __shared__ unsigned short As[2][64][64];
  __shared__ unsigned short Bs[2][64][64];
  const int tid  = threadIdx.x;
  const int wave = tid >> 5, lane = tid & 31;
  const int m0 = blockIdx.y * 64, n0 = blockIdx.x * 64;
  const long long zb = blockIdx.z;
  const float* Ab = A + zb * a_bs;
  const float* Bb = B + zb * b_bs;
  float*       Cb = C + zb * c_bs;

  const bool fullM = (m0 + 64 <= M);
  const bool fullN = (n0 + 64 <= N);

  auto loadTile = [&](int buf, int kk0) {
    const bool fullK = (kk0 + 64 <= K);
    // ---- stage A tile (64 rows x 64 k) ----
    if (fullK && fullM && a_cs == 1) {
      #pragma unroll
      for (int t = 0; t < 8; ++t) {
        int idx = tid + t * 128;        // 0..1023 vec4 slots
        int r = idx >> 4, c4 = idx & 15;
        float4 f = *reinterpret_cast<const float4*>(
            Ab + (long long)(m0 + r) * a_rs + kk0 + (c4 << 2));
        ushort4 u;
        u.x = f2bf(f.x); u.y = f2bf(f.y); u.z = f2bf(f.z); u.w = f2bf(f.w);
        *reinterpret_cast<ushort4*>(&As[buf][r][c4 << 2]) = u;
      }
    } else {
      for (int i = tid; i < 64 * 64; i += 128) {
        int r = i >> 6, c = i & 63;
        int gm = m0 + r, gk = kk0 + c;
        float v = (gm < M && gk < K) ? Ab[(long long)gm * a_rs + (long long)gk * a_cs] : 0.f;
        As[buf][r][c] = f2bf(v);
      }
    }
    // ---- stage B tile (64 k x 64 n) ----
    if (fullK && fullN && b_cs == 1) {
      #pragma unroll
      for (int t = 0; t < 8; ++t) {
        int idx = tid + t * 128;
        int r = idx >> 4, c4 = idx & 15;
        float4 f = *reinterpret_cast<const float4*>(
            Bb + (long long)(kk0 + r) * b_rs + n0 + (c4 << 2));
        ushort4 u;
        u.x = f2bf(f.x); u.y = f2bf(f.y); u.z = f2bf(f.z); u.w = f2bf(f.w);
        *reinterpret_cast<ushort4*>(&Bs[buf][r][c4 << 2]) = u;
      }
    } else if (fullK && fullN && b_rs == 1) {
      // transpose view (e.g. K^T in attention): contiguous along k
      #pragma unroll
      for (int t = 0; t < 8; ++t) {
        int idx = tid + t * 128;
        int n = idx >> 4, k4 = idx & 15;
        float4 f = *reinterpret_cast<const float4*>(
            Bb + (long long)(n0 + n) * b_cs + kk0 + (k4 << 2));
        Bs[buf][(k4 << 2) + 0][n] = f2bf(f.x);
        Bs[buf][(k4 << 2) + 1][n] = f2bf(f.y);
        Bs[buf][(k4 << 2) + 2][n] = f2bf(f.z);
        Bs[buf][(k4 << 2) + 3][n] = f2bf(f.w);
      }
    } else if (fullK && fullN) {        // generic strided but in-range
      for (int i = tid; i < 64 * 64; i += 128) {
        int r = i >> 6, c = i & 63;
        Bs[buf][r][c] = f2bf(Bb[(long long)(kk0 + r) * b_rs + (long long)(n0 + c) * b_cs]);
      }
    } else {
      for (int i = tid; i < 64 * 64; i += 128) {
        int r = i >> 6, c = i & 63;
        int gk = kk0 + r, gn = n0 + c;
        float v = (gk < K && gn < N) ? Bb[(long long)gk * b_rs + (long long)gn * b_cs] : 0.f;
        Bs[buf][r][c] = f2bf(v);
      }
    }
  };

  v8f acc[4] = {};
  const int mloc = (wave << 4) + (lane & 15);
  const int kb   = (lane & 16) ? 8 : 0;
  const int krow = (lane & 15) + ((lane & 16) ? 16 : 0);

  const int ntiles = (K + 63) >> 6;
  loadTile(0, 0);
  __syncthreads();

  for (int t = 0; t < ntiles; ++t) {
    if (t + 1 < ntiles) loadTile((t + 1) & 1, (t + 1) << 6);
    if (t + 2 < ntiles) {
      __builtin_prefetch(Ab + (long long)(m0 + (tid & 63)) * a_rs + (long long)((t + 2) << 6) * a_cs, 0, 0);
      __builtin_prefetch(Bb + (long long)((t + 2) << 6) * b_rs + (long long)n0 * b_cs, 0, 0);
    }
    const int cur = t & 1;
    #pragma unroll
    for (int ks = 0; ks < 64; ks += 32) {
      Frag16 af;
      #pragma unroll
      for (int i = 0; i < 16; ++i) {
        int kk = (i < 8) ? (kb + i) : (kb + 8 + i);
        af.u[i] = As[cur][mloc][ks + kk];
      }
      #pragma unroll
      for (int nt = 0; nt < 4; ++nt) {
        Frag16 bf;
        #pragma unroll
        for (int i = 0; i < 16; ++i) bf.u[i] = Bs[cur][ks + ((i < 8) ? (kb + i) : (kb + 8 + i))][(nt << 4) + (lane & 15)];
        acc[nt] = __builtin_amdgcn_wmma_f32_16x16x32_bf16(
            false, af.v, false, bf.v, (short)0, acc[nt], false, false);
      }
    }
    __syncthreads();
  }

  const int mrow = m0 + (wave << 4) + ((lane >> 4) << 3);
  const int ncol = lane & 15;
  if (fullM && fullN) {
    #pragma unroll
    for (int nt = 0; nt < 4; ++nt) {
      int gn = n0 + (nt << 4) + ncol;
      float bv = bias ? bias[gn] : 0.f;
      #pragma unroll
      for (int r = 0; r < 8; ++r) {
        float v = acc[nt][r] * scale + bv;
        if (act == 1) v = geluf(v);
        else if (act == 2) v = softplusf(v);
        long long idx = (long long)(mrow + r) * c_rs + gn;
        if (accum) v += Cb[idx];
        Cb[idx] = v;
      }
    }
  } else {
    #pragma unroll
    for (int nt = 0; nt < 4; ++nt) {
      int gn = n0 + (nt << 4) + ncol;
      if (gn >= N) continue;
      float bv = bias ? bias[gn] : 0.f;
      #pragma unroll
      for (int r = 0; r < 8; ++r) {
        int gm = mrow + r;
        if (gm >= M) continue;
        float v = acc[nt][r] * scale + bv;
        if (act == 1) v = geluf(v);
        else if (act == 2) v = softplusf(v);
        long long idx = (long long)gm * c_rs + gn;
        if (accum) v += Cb[idx];
        Cb[idx] = v;
      }
    }
  }
}

// ---------------- Small / scan kernels ----------------
__global__ void sentproj_kernel(const float* se, const float* W, const float* b, float* out) {
  int o = blockIdx.x * blockDim.x + threadIdx.x;
  if (o >= 768) return;
  float s = b[o];
  for (int i = 0; i < 768; ++i) s += se[i] * W[i * 768 + o];
  out[o] = s;
}

__global__ void embed_rope_kernel(const int* ids, const float* emb, const float* sv, float* x) {
  int idx = blockIdx.x * blockDim.x + threadIdx.x;
  if (idx >= S_LEN * 384) return;
  int s = idx / 384, j = idx % 384;
  long long tok = ids[s];
  float x1 = emb[tok * 768 + j] + sv[j];
  float x2 = emb[tok * 768 + 384 + j] + sv[384 + j];
  float inv = __expf(-(2.f * j / 768.f) * 9.210340371976184f); // ln(10000)
  float th = (float)s * inv;
  float c = __cosf(th), sn = __sinf(th);
  x[s * 768 + j]       = x1 * c - x2 * sn;
  x[s * 768 + 384 + j] = x2 * c + x1 * sn;
}

__global__ void cema_tsnorm_kernel(const float* alpha, float* x) {
  int d = blockIdx.x * blockDim.x + threadIdx.x;
  if (d >= 768) return;
  float a = 1.f / (1.f + __expf(-alpha[d]));
  float h = 0.f, sum = 0.f, sq = 0.f;
  for (int t = 0; t < S_LEN; ++t) {
    float v = x[t * 768 + d];
    h = a * v + (1.f - a) * h;
    sum += h; sq += h * h;
    float cnt = (float)(t + 1);
    float cm = sum / cnt;
    float cv = sq / cnt - cm * cm;
    x[t * 768 + d] = (h - cm) * rsqrtf(fmaxf(cv, 0.f) + EPSF);
  }
}

__global__ void mamba_conv_kernel(const float* xz, const float* cw, const float* cb, float* xc) {
  int idx = blockIdx.x * blockDim.x + threadIdx.x;
  if (idx >= S_LEN * D_IN) return;
  int c = idx % D_IN, t = idx / D_IN;
  float s = cb[c];
  #pragma unroll
  for (int j = 0; j < 4; ++j) {
    int tt = t - 3 + j;
    if (tt >= 0) s += xz[tt * 3072 + c] * cw[c * 4 + j];
  }
  xc[t * D_IN + c] = siluf(s);
}

// one thread per (channel, state); 16-lane shuffle reduction over states
__global__ void mamba_scan_kernel(const float* dt, const float* xdb, const float* xc,
                                  const float* xz, const float* A_log, const float* Dp,
                                  float* ymul) {
  int gtid = blockIdx.x * blockDim.x + threadIdx.x;
  int j = gtid & 15;
  int c = gtid >> 4;
  if (c >= D_IN) return;
  float Aj = -__expf(A_log[c * 16 + j]);
  float Dc = Dp[c];
  float h = 0.f;
  for (int t = 0; t < S_LEN; ++t) {
    float dtv = dt[t * D_IN + c];
    float bt  = xdb[t * 800 + 768 + j];
    float ct  = xdb[t * 800 + 784 + j];
    float xt  = xc[t * D_IN + c];
    h = __expf(dtv * Aj) * h + dtv * xt * bt;
    float p = h * ct;
    p += __shfl_xor(p, 1, 32);
    p += __shfl_xor(p, 2, 32);
    p += __shfl_xor(p, 4, 32);
    p += __shfl_xor(p, 8, 32);
    if (j == 0) {
      float z = xz[t * 3072 + D_IN + c];
      ymul[t * D_IN + c] = (p + Dc * xt) * siluf(z);
    }
  }
}

// out = LN(x + y) with affine; y may be null. One block per row (D=768).
__global__ void add_ln_kernel(const float* x, const float* y, const float* g,
                              const float* b, float* out) {
  __shared__ float r1[256], r2[256];
  int row = blockIdx.x, tid = threadIdx.x;
  float v[3]; float s = 0.f, sq = 0.f;
  #pragma unroll
  for (int i = 0; i < 3; ++i) {
    int d = tid + i * 256;
    float t = x[row * 768 + d];
    if (y) t += y[row * 768 + d];
    v[i] = t; s += t; sq += t * t;
  }
  r1[tid] = s; r2[tid] = sq; __syncthreads();
  for (int st = 128; st > 0; st >>= 1) {
    if (tid < st) { r1[tid] += r1[tid + st]; r2[tid] += r2[tid + st]; }
    __syncthreads();
  }
  float mean = r1[0] / 768.f;
  float var  = r2[0] / 768.f - mean * mean;
  float rs = rsqrtf(fmaxf(var, 0.f) + EPSF);
  #pragma unroll
  for (int i = 0; i < 3; ++i) {
    int d = tid + i * 256;
    out[row * 768 + d] = (v[i] - mean) * rs * g[d] + b[d];
  }
}

// row softmax over 1024 cols with mask; one block per (head,row)
__global__ void softmax_kernel(float* sc, const float* mask) {
  __shared__ float red[256];
  long long row = blockIdx.x;
  float* p = sc + row * 1024;
  int tid = threadIdx.x;
  float v[4]; float mx = -1e30f;
  #pragma unroll
  for (int i = 0; i < 4; ++i) {
    int col = tid + i * 256;
    float t = (mask[col] > 0.f) ? p[col] : -1e9f;
    v[i] = t; mx = fmaxf(mx, t);
  }
  red[tid] = mx; __syncthreads();
  for (int st = 128; st > 0; st >>= 1) { if (tid < st) red[tid] = fmaxf(red[tid], red[tid + st]); __syncthreads(); }
  mx = red[0]; __syncthreads();
  float s = 0.f;
  #pragma unroll
  for (int i = 0; i < 4; ++i) { v[i] = __expf(v[i] - mx); s += v[i]; }
  red[tid] = s; __syncthreads();
  for (int st = 128; st > 0; st >>= 1) { if (tid < st) red[tid] += red[tid + st]; __syncthreads(); }
  float inv = 1.f / red[0];
  #pragma unroll
  for (int i = 0; i < 4; ++i) p[tid + i * 256] = v[i] * inv;
}

__global__ void silu_kernel(const float* in, float* out, int n) {
  int i = blockIdx.x * blockDim.x + threadIdx.x;
  if (i < n) out[i] = siluf(in[i]);
}
__global__ void add_inplace_kernel(float* x, const float* y, int n) {
  int i = blockIdx.x * blockDim.x + threadIdx.x;
  if (i < n) x[i] += y[i];
}

// phi[s, g*Din + d] = silu(x[s,d] - grid[g]); grid = linspace(-2,2,4)
__global__ void phi_kernel(const float* x, float* phi, int Din, int n) {
  int idx = blockIdx.x * blockDim.x + threadIdx.x;
  if (idx >= n) return;
  int d = idx % Din;
  int g = (idx / Din) & 3;
  int s = idx / (Din * 4);
  float gv = -2.f + g * (4.f / 3.f);
  phi[idx] = siluf(x[s * Din + d] - gv);
}

__global__ void gp_feats_kernel(const float* h, const float* Z, const float* logls, float* feats) {
  int idx = blockIdx.x * blockDim.x + threadIdx.x;
  if (idx >= S_LEN * 20) return;
  int f = idx % 20, s = idx / 20;
  int l = f / 10, n = f % 10;
  const float* z  = Z + (l * 10 + n) * 768;
  const float* hp = h + s * 768;
  float d2 = 0.f;
  for (int d = 0; d < 768; ++d) { float df = hp[d] - z[d]; d2 += df * df; }
  float ls2 = __expf(2.f * logls[l]);
  feats[s * 20 + f] = __expf(-d2 / (2.f * ls2));
}

static inline void gemm(hipStream_t st, const float* A, const float* B, float* C,
    int M, int N, int K,
    long long ars, long long acs, long long abs_,
    long long brs, long long bcs, long long bbs,
    long long crs, long long cbs,
    const float* bias, float scale, int act, int accum, int batch) {
  dim3 grid((N + 63) / 64, (M + 63) / 64, batch);
  gemm_wmma_kernel<<<grid, 128, 0, st>>>(A, B, C, M, N, K,
      ars, acs, abs_, brs, bcs, bbs, crs, cbs, bias, scale, act, accum);
}

extern "C" void kernel_launch(void* const* d_in, const int* in_sizes, int n_in,
                              void* d_out, int out_size, void* d_ws, size_t ws_size,
                              hipStream_t stream) {
  const int*   ids   = (const int*)d_in[0];
  const float* mask  = (const float*)d_in[1];
  const float* semb  = (const float*)d_in[2];
  const float* etab  = (const float*)d_in[3];
  const float* sentW = (const float*)d_in[4];
  const float* sentb = (const float*)d_in[5];
  const float* alpha = (const float*)d_in[6];
  const float* inW   = (const float*)d_in[7];
  const float* convW = (const float*)d_in[8];
  const float* convB = (const float*)d_in[9];
  const float* xW    = (const float*)d_in[10];
  const float* dtW   = (const float*)d_in[11];
  const float* dtB   = (const float*)d_in[12];
  const float* Alog  = (const float*)d_in[13];
  const float* Dp    = (const float*)d_in[14];
  const float* outW  = (const float*)d_in[15];
  const float* projW = (const float*)d_in[16];
  const float* projb = (const float*)d_in[17];
  const float* qkvW  = (const float*)d_in[18];
  const float* qkvb  = (const float*)d_in[19];
  const float* oW    = (const float*)d_in[20];
  const float* ob    = (const float*)d_in[21];
  const float* g1    = (const float*)d_in[22];
  const float* b1    = (const float*)d_in[23];
  const float* g2    = (const float*)d_in[24];
  const float* b2    = (const float*)d_in[25];
  const float* f1W   = (const float*)d_in[26];
  const float* f1b   = (const float*)d_in[27];
  const float* f2W   = (const float*)d_in[28];
  const float* f2b   = (const float*)d_in[29];
  const float* k1b   = (const float*)d_in[30];
  const float* k1s   = (const float*)d_in[31];
  const float* k2b   = (const float*)d_in[32];
  const float* k2s   = (const float*)d_in[33];
  const float* fg    = (const float*)d_in[34];
  const float* fb    = (const float*)d_in[35];
  const float* gZ    = (const float*)d_in[36];
  const float* gls   = (const float*)d_in[37];
  const float* uW    = (const float*)d_in[38];
  const float* ub    = (const float*)d_in[39];

  float* hout = (float*)d_out;                 // (1024,768)
  float* uout = hout + S_LEN * D_MODEL;        // (1024,50257)

  float* W = (float*)d_ws;
  float* x      = W;                           // 1024x768
  float* bufA   = x + 786432;                  // 1024x3072 (xz/qkv/ff1/k1)
  float* bufB   = bufA + 3145728;              // 1024x1536 (xc)
  float* bufC   = bufB + 1572864;              // 1024x800  (xdb)
  float* bufD   = bufC + 819200;               // 1024x1536 (dt)
  float* bufE   = bufD + 1572864;              // 1024x1536 (ymul)
  float* bufF   = bufE + 1572864;              // 1024x768
  float* bufG   = bufF + 786432;               // 1024x768
  float* scores = bufG + 786432;               // 12x1024x1024, aliased with phi
  float* phi    = scores;                      // up to 1024x12288
  float* sentv  = scores + 12582912;           // 768
  float* feats  = sentv + 768;                 // 1024x20

  sentproj_kernel<<<3, 256, 0, stream>>>(semb, sentW, sentb, sentv);
  embed_rope_kernel<<<(S_LEN * 384 + 255) / 256, 256, 0, stream>>>(ids, etab, sentv, x);
  cema_tsnorm_kernel<<<3, 256, 0, stream>>>(alpha, x);

  for (int l = 0; l < 2; ++l) {
    // ---- Mamba ----
    gemm(stream, x, inW + (long long)l * 768 * 3072, bufA, 1024, 3072, 768,
         768, 1, 0, 3072, 1, 0, 3072, 0, nullptr, 1.f, 0, 0, 1);
    mamba_conv_kernel<<<(S_LEN * D_IN + 255) / 256, 256, 0, stream>>>(
        bufA, convW + l * 1536 * 4, convB + l * 1536, bufB);
    gemm(stream, bufB, xW + (long long)l * 1536 * 800, bufC, 1024, 800, 1536,
         1536, 1, 0, 800, 1, 0, 800, 0, nullptr, 1.f, 0, 0, 1);
    gemm(stream, bufC, dtW + (long long)l * 768 * 1536, bufD, 1024, 1536, 768,
         800, 1, 0, 1536, 1, 0, 1536, 0, dtB + l * 1536, 1.f, 2, 0, 1);
    mamba_scan_kernel<<<96, 256, 0, stream>>>(bufD, bufC, bufB, bufA,
        Alog + l * 1536 * 16, Dp + l * 1536, bufE);
    gemm(stream, bufE, outW + (long long)l * 1536 * 768, bufF, 1024, 768, 1536,
         1536, 1, 0, 768, 1, 0, 768, 0, nullptr, 1.f, 0, 0, 1);
    gemm(stream, bufF, projW + (long long)l * 768 * 768, x, 1024, 768, 768,
         768, 1, 0, 768, 1, 0, 768, 0, projb + l * 768, 1.f, 0, 0, 1);

    // ---- Attention ----
    gemm(stream, x, qkvW + (long long)l * 768 * 2304, bufA, 1024, 2304, 768,
         768, 1, 0, 2304, 1, 0, 2304, 0, qkvb + l * 2304, 1.f, 0, 0, 1);
    gemm(stream, bufA, bufA + 768, scores, 1024, 1024, 64,
         2304, 1, 64, 1, 2304, 64, 1024, 1048576, nullptr, 0.125f, 0, 0, 12);
    softmax_kernel<<<12 * 1024, 256, 0, stream>>>(scores, mask);
    gemm(stream, scores, bufA + 1536, bufG, 1024, 64, 1024,
         1024, 1, 1048576, 2304, 1, 64, 768, 64, nullptr, 1.f, 0, 0, 12);
    gemm(stream, bufG, oW + (long long)l * 768 * 768, bufF, 1024, 768, 768,
         768, 1, 0, 768, 1, 0, 768, 0, ob + l * 768, 1.f, 0, 0, 1);
    add_ln_kernel<<<1024, 256, 0, stream>>>(x, bufF, g1 + l * 768, b1 + l * 768, x);
    gemm(stream, x, f1W + (long long)l * 768 * 3072, bufA, 1024, 3072, 768,
         768, 1, 0, 3072, 1, 0, 3072, 0, f1b + l * 3072, 1.f, 1, 0, 1);
    gemm(stream, bufA, f2W + (long long)l * 3072 * 768, bufF, 1024, 768, 3072,
         3072, 1, 0, 768, 1, 0, 768, 0, f2b + l * 768, 1.f, 0, 0, 1);
    add_ln_kernel<<<1024, 256, 0, stream>>>(x, bufF, g2 + l * 768, b2 + l * 768, x);

    // ---- KAN (double) ----
    silu_kernel<<<(786432 + 255) / 256, 256, 0, stream>>>(x, bufG, 786432);
    gemm(stream, bufG, k1b + (long long)l * 768 * 3072, bufA, 1024, 3072, 768,
         768, 1, 0, 3072, 1, 0, 3072, 0, nullptr, 1.f, 0, 0, 1);
    phi_kernel<<<(1024 * 4 * 768 + 255) / 256, 256, 0, stream>>>(x, phi, 768, 1024 * 4 * 768);
    gemm(stream, phi, k1s + (long long)l * 4 * 768 * 3072, bufA, 1024, 3072, 3072,
         3072, 1, 0, 3072, 1, 0, 3072, 0, nullptr, 1.f, 0, 1, 1);
    phi_kernel<<<(1024 * 4 * 3072 + 255) / 256, 256, 0, stream>>>(bufA, phi, 3072, 1024 * 4 * 3072);
    silu_kernel<<<(3145728 + 255) / 256, 256, 0, stream>>>(bufA, bufA, 3145728);
    gemm(stream, bufA, k2b + (long long)l * 3072 * 768, bufF, 1024, 768, 3072,
         3072, 1, 0, 768, 1, 0, 768, 0, nullptr, 1.f, 0, 0, 1);
    gemm(stream, phi, k2s + (long long)l * 4 * 3072 * 768, bufF, 1024, 768, 12288,
         12288, 1, 0, 768, 1, 0, 768, 0, nullptr, 1.f, 0, 1, 1);
    add_inplace_kernel<<<(786432 + 255) / 256, 256, 0, stream>>>(x, bufF, 786432);
  }

  add_ln_kernel<<<1024, 256, 0, stream>>>(x, nullptr, fg, fb, hout);
  gp_feats_kernel<<<(S_LEN * 20 + 255) / 256, 256, 0, stream>>>(hout, gZ, gls, feats);
  gemm(stream, feats, uW, uout, 1024, VOCAB, 20,
       20, 1, 0, VOCAB, 1, 0, VOCAB, 0, ub, 1.f, 2, 0, 1);
}
